// HANSQL_47682726920408
// MI455X (gfx1250) — matrix operand order repaired and checked by reading, not compiled, not run on hardware
//
#include <hip/hip_runtime.h>
#include <cstddef>
#include <cstdint>

#define GN_Q 40000
#define GN_T 10000
#define GN_C 50000
#define GN   (GN_Q + GN_T + GN_C)
#define GD   256
#define GH   8
#define GDK  32
#define GLAY 2
#define NMAX 50000

typedef _Float16 v16h __attribute__((ext_vector_type(16)));
typedef _Float16 v8h  __attribute__((ext_vector_type(8)));
typedef float    v8f  __attribute__((ext_vector_type(8)));
typedef int      v4i  __attribute__((vector_size(4 * sizeof(int))));
typedef __attribute__((address_space(1))) v4i* gv4i_p;   // global
typedef __attribute__((address_space(3))) v4i* lv4i_p;   // LDS

// Tile geometry
#define BM     128
#define BN     64
#define BK     32
#define LDPB   40    // B LDS row pitch in halfs (80 B: bank spread + 16B align)
#define LDPA   36    // A LDS row pitch in floats (144 B: bank spread + 16B align)

#if __has_builtin(__builtin_amdgcn_global_load_async_to_lds_b128)
#define HAVE_ASYNC_LDS 1
#else
#define HAVE_ASYNC_LDS 0
#endif

// ---------------------------------------------------------------------------
// WMMA GEMM: C[M,N] = op(A[M,K] @ B[K,N] (+bias) (+res)), op = relu or id.
// Block = 256 threads = 8 waves. Each wave owns a 16x64 strip (4 wmma tiles,
// one shared A fragment). A tile staged fp32 global->LDS via async-to-LDS
// (ASYNCcnt path) when available; B tile transposed+converted to f16 in LDS.
// Fragments: A = 4x ds_load_b128 fp32 + v_cvt_pk_f16_f32; B = 2x ds_load_b128.
// All 5 fragments preloaded, then 4 back-to-back v_wmma_f32_16x16x32_f16.
// Requires: M % 16 == 0, N % 64 == 0, K % 32 == 0 (true for this model).
// ---------------------------------------------------------------------------
template <bool BIAS, bool RES, bool RELU>
__global__ __launch_bounds__(256) void hgt_wmma_gemm(
    const float* __restrict__ A, const float* __restrict__ B,
    const float* __restrict__ bias, const float* __restrict__ res,
    float* __restrict__ C, int M, int Ncols, int K)
{
    __shared__ float     Af[BM * LDPA];   // A tile fp32 [128][32] (+pad) 18 KB
    __shared__ _Float16  Bt[BN * LDPB];   // B tile f16, transposed [64][32] 5 KB

    const int tid  = threadIdx.x;
    const int lane = tid & 31;
    const int wave = tid >> 5;

    const int rowBase0 = blockIdx.y * BM;
    const int colBase0 = blockIdx.x * BN;
    const int waveRow  = wave * 16;
    const bool active  = (rowBase0 + waveRow) < M;   // uniform per wave

    // B staging assignment (coalesced rows of B)
    const int bn = tid & 63;                   // col 0..63
    const int bk = (tid >> 6) * 8;             // k 0,8,16,24

    // Fragment addressing (wave32 WMMA layouts)
    const int fm    = lane & 15;               // M (A) or N (B) within tile
    const int aKoff = (lane >> 4) << 4;        // A: k-base 0 or 16
    const int bKoff = (lane >> 4) << 4;        // B: k-base 0 or 16

    v8f acc[4] = {};

    for (int kk = 0; kk < K; kk += BK) {
        __syncthreads();   // protect previous iteration's LDS reads

        // ---- stage A tile: 128x32 fp32, 16B chunks, 4 per thread ----
        #pragma unroll
        for (int c = 0; c < 4; ++c) {
            const int chunk = tid + 256 * c;            // 0..1023
            const int r  = chunk >> 3;                  // row 0..127
            const int c4 = (chunk & 7) * 4;             // col 0,4,..,28
            const int sRow = (rowBase0 + r < M) ? (rowBase0 + r) : (M - 1);
            const float* gp = A + (size_t)sRow * K + kk + c4;
            float* lp = &Af[r * LDPA + c4];
#if HAVE_ASYNC_LDS
            __builtin_amdgcn_global_load_async_to_lds_b128(
                (gv4i_p)const_cast<float*>(gp), (lv4i_p)lp, 0, 0);
#else
            *(float4*)lp = *(const float4*)gp;
#endif
        }

        // ---- stage B tile transposed+converted (overlaps with async A) ----
        {
            const float* bp = B + (size_t)(kk + bk) * Ncols + colBase0 + bn;
            #pragma unroll
            for (int r = 0; r < 8; ++r)
                Bt[bn * LDPB + bk + r] = (_Float16)bp[(size_t)r * Ncols];
        }

#if HAVE_ASYNC_LDS
#if __has_builtin(__builtin_amdgcn_s_wait_asynccnt)
        __builtin_amdgcn_s_wait_asynccnt(0);
#else
        asm volatile("s_wait_asynccnt 0" ::: "memory");
#endif
#endif
        __syncthreads();

        if (active) {
            // A fragment: 4x 16B fp32 runs -> convert to f16
            const float* arow = &Af[(waveRow + fm) * LDPA + aKoff];
            const float4 fa0 = *(const float4*)(arow);
            const float4 fa1 = *(const float4*)(arow + 4);
            const float4 fa2 = *(const float4*)(arow + 8);
            const float4 fa3 = *(const float4*)(arow + 12);
            v16h a;
            a[0]  = (_Float16)fa0.x; a[1]  = (_Float16)fa0.y;
            a[2]  = (_Float16)fa0.z; a[3]  = (_Float16)fa0.w;
            a[4]  = (_Float16)fa1.x; a[5]  = (_Float16)fa1.y;
            a[6]  = (_Float16)fa1.z; a[7]  = (_Float16)fa1.w;
            a[8]  = (_Float16)fa2.x; a[9]  = (_Float16)fa2.y;
            a[10] = (_Float16)fa2.z; a[11] = (_Float16)fa2.w;
            a[12] = (_Float16)fa3.x; a[13] = (_Float16)fa3.y;
            a[14] = (_Float16)fa3.z; a[15] = (_Float16)fa3.w;

            // Preload all 4 B fragments (8x ds_load_b128 as one clause)
            v16h bfr[4];
            #pragma unroll
            for (int nt = 0; nt < 4; ++nt) {
                const _Float16* bcol = &Bt[(nt * 16 + fm) * LDPB + bKoff];
                const v8h b0 = *(const v8h*)(bcol);
                const v8h b1 = *(const v8h*)(bcol + 8);
                #pragma unroll
                for (int w = 0; w < 8; ++w) { bfr[nt][w] = b0[w]; bfr[nt][8 + w] = b1[w]; }
            }
            // 4 back-to-back WMMAs sharing the A fragment
            #pragma unroll
            for (int nt = 0; nt < 4; ++nt)
                acc[nt] = __builtin_amdgcn_wmma_f32_16x16x32_f16(
                    false, a, false, bfr[nt], (short)0, acc[nt], false, false);
        }
    }

    if (!active) return;

    // Epilogue: C lane map: N = lane&15, VGPR r -> M = r + (lane>=16 ? 8 : 0)
    const int rb = rowBase0 + waveRow + ((lane >> 4) << 3);
    #pragma unroll
    for (int nt = 0; nt < 4; ++nt) {
        const int col = colBase0 + nt * 16 + fm;
        const float bv = BIAS ? bias[col] : 0.0f;
        #pragma unroll
        for (int r = 0; r < 8; ++r) {
            const size_t idx = (size_t)(rb + r) * Ncols + col;
            float v = acc[nt][r] + bv;
            if (RES)  v += res[idx];
            if (RELU) v = fmaxf(v, 0.0f);
            C[idx] = v;
        }
    }
}

// ---------------------------------------------------------------------------
// Edge attention: 8 contiguous edges per dst node (dst = repeat(arange, 8)).
// ---------------------------------------------------------------------------
__global__ __launch_bounds__(256) void hgt_attn(
    const float* __restrict__ q,
    const float* __restrict__ k0, const float* __restrict__ v0, const int* __restrict__ s0,
    const float* __restrict__ k1, const float* __restrict__ v1, const int* __restrict__ s1,
    float* __restrict__ o, int n, int two)
{
    const int t = blockIdx.x * blockDim.x + threadIdx.x;
    if (t >= n * GH) return;
    const int node = t >> 3;
    const int head = t & 7;

    const float4* qp = (const float4*)(q + (size_t)node * GD + head * GDK);
    float4 qr[8];
    #pragma unroll
    for (int j = 0; j < 8; ++j) qr[j] = qp[j];

    float wv[32];
    #pragma unroll
    for (int d = 0; d < 32; ++d) wv[d] = 0.0f;
    float z = 0.0f;

    const int ng = two ? 2 : 1;
    for (int g = 0; g < ng; ++g) {
        const float* kp_ = g ? k1 : k0;
        const float* vp_ = g ? v1 : v0;
        const int*   sp_ = g ? s1 : s0;
        for (int e = 0; e < 8; ++e) {
            const int s = sp_[(size_t)node * 8 + e];
            const float4* kp = (const float4*)(kp_ + (size_t)s * GD + head * GDK);
            float dot = 0.0f;
            #pragma unroll
            for (int j = 0; j < 8; ++j) {
                float4 kf = kp[j];
                dot += qr[j].x * kf.x + qr[j].y * kf.y + qr[j].z * kf.z + qr[j].w * kf.w;
            }
            const float w = __expf(fminf(fmaxf(dot * 0.17677669529663687f, -5.0f), 5.0f));
            z += w;
            const float4* vp = (const float4*)(vp_ + (size_t)s * GD + head * GDK);
            #pragma unroll
            for (int j = 0; j < 8; ++j) {
                float4 vf = vp[j];
                wv[4 * j + 0] += w * vf.x;
                wv[4 * j + 1] += w * vf.y;
                wv[4 * j + 2] += w * vf.z;
                wv[4 * j + 3] += w * vf.w;
            }
        }
    }

    const float rz = 1.0f / (z + 1e-9f);
    float4* op = (float4*)(o + (size_t)node * GD + head * GDK);
    #pragma unroll
    for (int j = 0; j < 8; ++j) {
        float4 r;
        r.x = wv[4 * j + 0] * rz; r.y = wv[4 * j + 1] * rz;
        r.z = wv[4 * j + 2] * rz; r.w = wv[4 * j + 3] * rz;
        op[j] = r;
    }
}

// ---------------------------------------------------------------------------
// Row LayerNorm over 256 columns, one wave32 per row.
// out = (accum ? out : 0) + scale * (LN(t)*g + b)
// ---------------------------------------------------------------------------
__global__ __launch_bounds__(256) void hgt_ln(
    const float* __restrict__ t, const float* __restrict__ g, const float* __restrict__ b,
    float* __restrict__ out, int n, float scale, int accum)
{
    const int wave = threadIdx.x >> 5;
    const int lane = threadIdx.x & 31;
    const int row  = blockIdx.x * 8 + wave;
    if (row >= n) return;

    const float* tp = t + (size_t)row * GD;
    float xv[8];
    float s = 0.0f;
    #pragma unroll
    for (int j = 0; j < 8; ++j) { xv[j] = tp[j * 32 + lane]; s += xv[j]; }
    #pragma unroll
    for (int m = 16; m >= 1; m >>= 1) s += __shfl_xor(s, m, 32);
    const float mean = s * (1.0f / 256.0f);

    float var = 0.0f;
    #pragma unroll
    for (int j = 0; j < 8; ++j) { const float d = xv[j] - mean; var += d * d; }
    #pragma unroll
    for (int m = 16; m >= 1; m >>= 1) var += __shfl_xor(var, m, 32);
    const float rstd = rsqrtf(var * (1.0f / 256.0f) + 1e-5f);

    float* op = out + (size_t)row * GD;
    #pragma unroll
    for (int j = 0; j < 8; ++j) {
        const int col = j * 32 + lane;
        const float r = (xv[j] - mean) * rstd * g[col] + b[col];
        if (accum) op[col] += scale * r;
        else       op[col]  = scale * r;
    }
}

// ---------------------------------------------------------------------------
// Host-side launchers
// ---------------------------------------------------------------------------
static inline void launch_gemm(const float* A, const float* B, const float* bias,
                               const float* res, float* C, int M, int Ncols, int K,
                               bool useBias, bool useRes, bool useRelu, hipStream_t s)
{
    dim3 grid(Ncols / BN, (M + BM - 1) / BM);
    if (useBias && useRes)
        hgt_wmma_gemm<true,  true,  false><<<grid, 256, 0, s>>>(A, B, bias, res, C, M, Ncols, K);
    else if (useBias && useRelu)
        hgt_wmma_gemm<true,  false, true ><<<grid, 256, 0, s>>>(A, B, bias, res, C, M, Ncols, K);
    else if (useBias)
        hgt_wmma_gemm<true,  false, false><<<grid, 256, 0, s>>>(A, B, bias, res, C, M, Ncols, K);
    else
        hgt_wmma_gemm<false, false, false><<<grid, 256, 0, s>>>(A, B, bias, res, C, M, Ncols, K);
}

static inline void launch_ln(const float* t, const float* g, const float* b, float* out,
                             int n, float scale, int accum, hipStream_t s)
{
    hgt_ln<<<(n + 7) / 8, 256, 0, s>>>(t, g, b, out, n, scale, accum);
}

extern "C" void kernel_launch(void* const* d_in, const int* in_sizes, int n_in,
                              void* d_out, int out_size, void* d_ws, size_t ws_size,
                              hipStream_t stream)
{
    const float* x0  = (const float*)d_in[0];
    const float* Wq  = (const float*)d_in[1];
    const float* bq  = (const float*)d_in[2];
    const float* Wk  = (const float*)d_in[3];
    const float* Wv  = (const float*)d_in[4];
    const float* Wo  = (const float*)d_in[5];
    const float* bo  = (const float*)d_in[6];
    const float* g1  = (const float*)d_in[7];
    const float* b1l = (const float*)d_in[8];
    const float* W1  = (const float*)d_in[9];
    const float* bf1 = (const float*)d_in[10];
    const float* W2  = (const float*)d_in[11];
    const float* bf2 = (const float*)d_in[12];
    const float* g2  = (const float*)d_in[13];
    const float* b2l = (const float*)d_in[14];
    // dict order per type: hs, hd, ss, sd, ds, dd (dst arrays unused: contiguous)
    const int* hs[3]; const int* ss[3]; const int* ds[3];
    for (int i = 0; i < 3; ++i) {
        hs[i] = (const int*)d_in[15 + 6 * i + 0];
        ss[i] = (const int*)d_in[15 + 6 * i + 2];
        ds[i] = (const int*)d_in[15 + 6 * i + 4];
    }

    // Workspace layout (floats)
    float* ws = (float*)d_ws;
    float* xc = ws;                                 // GN x GD   (inter-layer x)
    float* qb = xc + (size_t)GN * GD;               // GN x GD
    float* kb = qb + (size_t)GN * GD;               // GN x GD
    float* vb = kb + (size_t)GN * GD;               // GN x GD
    float* ob = vb + (size_t)GN * GD;               // NMAX x GD (attention out)
    float* tb = ob + (size_t)NMAX * GD;             // NMAX x GD (pre-LN temp)
    float* hb = tb + (size_t)NMAX * GD;             // NMAX x GD (post-LN h)
    float* ub = hb + (size_t)NMAX * GD;             // NMAX x 4GD (FFN hidden)
    (void)ws_size; (void)in_sizes; (void)n_in; (void)out_size;

    const int sizes[3] = {GN_Q, GN_T, GN_C};
    const int bases[3] = {0, GN_Q, GN_Q + GN_T};
    const int ends[3]  = {1, 2, 0};

    for (int l = 0; l < GLAY; ++l) {
        const float* xin  = (l == 0) ? x0 : xc;
        float*       xout = (l == GLAY - 1) ? (float*)d_out : xc;

        // Q/K/V projections for all three node types
        for (int i = 0; i < 3; ++i) {
            const size_t w    = (size_t)(l * 3 + i) * GD * GD;
            const size_t bofs = (size_t)(l * 3 + i) * GD;
            const size_t base = (size_t)bases[i] * GD;
            const int M = sizes[i];
            launch_gemm(xin + base, Wq + w, bq + bofs, nullptr, qb + base, M, GD, GD, true,  false, false, stream);
            launch_gemm(xin + base, Wk + w, nullptr,   nullptr, kb + base, M, GD, GD, false, false, false, stream);
            launch_gemm(xin + base, Wv + w, nullptr,   nullptr, vb + base, M, GD, GD, false, false, false, stream);
        }

        // Per-type: homo metapath, then het metapath; each through Wo+FFN block
        for (int i = 0; i < 3; ++i) {
            const int    M     = sizes[i];
            const int    j     = ends[i];
            const size_t base  = (size_t)bases[i] * GD;
            const size_t basej = (size_t)bases[j] * GD;
            const size_t w     = (size_t)(l * 3 + i) * GD * GD;
            const size_t w4    = (size_t)(l * 3 + i) * GD * 4 * GD;
            const size_t bofs  = (size_t)(l * 3 + i) * GD;
            const size_t b4    = (size_t)(l * 3 + i) * 4 * GD;
            const int gridA = (M * GH + 255) / 256;

            for (int path = 0; path < 2; ++path) {
                if (path == 0) {
                    hgt_attn<<<gridA, 256, 0, stream>>>(
                        qb + base, kb + base, vb + base, hs[i],
                        nullptr, nullptr, nullptr, ob, M, 0);
                } else {
                    hgt_attn<<<gridA, 256, 0, stream>>>(
                        qb + base, kb + basej, vb + basej, ss[i],
                        kb + base, vb + base, ds[i], ob, M, 1);
                }
                // t = x + o @ Wo + bo ; h = LN1(t)
                launch_gemm(ob, Wo + w, bo + bofs, xin + base, tb, M, GD, GD, true, true, false, stream);
                launch_ln(tb, g1 + bofs, b1l + bofs, hb, M, 1.0f, 0, stream);
                // u = relu(h @ W1 + bf1) ; t = h + u @ W2 + bf2
                launch_gemm(hb, W1 + w4, bf1 + b4, nullptr, ub, M, 4 * GD, GD, true, false, true, stream);
                launch_gemm(ub, W2 + w4, bf2 + bofs, hb, tb, M, GD, 4 * GD, true, true, false, stream);
                // xout_slice (+)= 0.5 * LN2(t)   (mean over the 2 metapaths)
                launch_ln(tb, g2 + bofs, b2l + bofs, xout + base, M, 0.5f, path, stream);
            }
        }
    }
}
